// MultiHeadAttentionLayer_80290118632232
// MI455X (gfx1250) — matrix operand order repaired and checked by reading, not compiled, not run on hardware
//
#include <hip/hip_runtime.h>
#include <hip/hip_bf16.h>
#include <cstdint>

#define E_DIM   1024
#define H_DIM   64
#define N_HEADS 16
#define BATCH   4
#define SEQ     2048

typedef __attribute__((ext_vector_type(16))) _Float16 v16h;
typedef __attribute__((ext_vector_type(8)))  float    v8f;
typedef __attribute__((__vector_size__(16))) int      v4i;

union U4H8 { uint4 u; _Float16 h[8]; };
union U2H4 { uint2 u; _Float16 h[4]; };

#if __has_builtin(__builtin_amdgcn_global_load_async_to_lds_b128)
#define HAVE_ASYNC_LDS 1
#else
#define HAVE_ASYNC_LDS 0
#endif

// 16B global -> LDS copy. Async (no data VGPRs, ASYNCcnt) when available.
__device__ __forceinline__ void copy_b128_to_lds(const _Float16* g, _Float16* l) {
#if HAVE_ASYNC_LDS
    auto gp = reinterpret_cast<v4i __attribute__((address_space(1)))*>(
                  reinterpret_cast<uintptr_t>(g));
    auto lp = reinterpret_cast<v4i __attribute__((address_space(3)))*>(
                  reinterpret_cast<uintptr_t>(l));
    __builtin_amdgcn_global_load_async_to_lds_b128(gp, lp, 0, 0);
#else
    *reinterpret_cast<uint4*>(l) = *reinterpret_cast<const uint4*>(g);
#endif
}

__device__ __forceinline__ void wait_async_copies() {
#if HAVE_ASYNC_LDS
#if __has_builtin(__builtin_amdgcn_s_wait_asynccnt)
    __builtin_amdgcn_s_wait_asynccnt(0);
#else
    asm volatile("s_wait_asynccnt 0x0" ::: "memory");
#endif
#endif
}

// ---------------------------------------------------------------------------
// fp32 -> f16 conversion, 4 elements/thread (grid-stride)
// ---------------------------------------------------------------------------
__global__ void cvt_f32_f16_kernel(const float4* __restrict__ src,
                                   uint2* __restrict__ dst, int n4) {
    int i = blockIdx.x * blockDim.x + threadIdx.x;
    int stride = gridDim.x * blockDim.x;
    for (; i < n4; i += stride) {
        float4 f = src[i];
        U2H4 t;
        t.h[0] = (_Float16)f.x; t.h[1] = (_Float16)f.y;
        t.h[2] = (_Float16)f.z; t.h[3] = (_Float16)f.w;
        dst[i] = t.u;
    }
}

// ---------------------------------------------------------------------------
// 16-bit WMMA lane layouts used by this code (wave32; A/B k order is its own contiguous choice):
//   A (16x32, MxK): lane L -> M=L&15, half=L>>4; elem e -> K=e+16*half
//   B (32x16, KxN): lane L -> N=L&15, half=L>>4; elem e -> K=e+16*half
//   C/D (16x16 f32): lane L, vgpr r -> M=r+8*(L>>4), N=L&15
// ---------------------------------------------------------------------------
__device__ __forceinline__ int a_frag_k(int e, int hf) {
    return 16 * hf + e;
}

__device__ __forceinline__ v8f wmma_f16(v16h a, v16h b, v8f c) {
    return __builtin_amdgcn_wmma_f32_16x16x32_f16(false, a, false, b,
                                                  (short)0, c, false, false);
}

// ---------------------------------------------------------------------------
// QKV projection: per (b,h): Q/K/V[2048,64] = X[2048,1024] @ W[1024,64] + bias
// grid: (SEQ/256, BATCH*N_HEADS, 3), block: 256 (8 waves, 32 rows x 64 cols each)
// A tile: async DMA global->LDS (double-buffered, no staging VGPRs).
// B tile: 1 uint4/thread register path (needed for the transpose scatter).
// ---------------------------------------------------------------------------
__global__ __launch_bounds__(256, 1) void qkv_proj_kernel(
    const _Float16* __restrict__ X,                                  // [B,S,E]
    const _Float16* __restrict__ Wq, const _Float16* __restrict__ Wk,
    const _Float16* __restrict__ Wv,                                 // [H,E,D]
    const float* __restrict__ bq, const float* __restrict__ bk,
    const float* __restrict__ bv,                                    // [H,D]
    _Float16* __restrict__ Q, _Float16* __restrict__ K,
    _Float16* __restrict__ V)                                        // [B,H,S,D]
{
    int tid = threadIdx.x, wave = tid >> 5, lane = tid & 31;
    int srow0 = blockIdx.x * 256;
    int b = blockIdx.y / N_HEADS, h = blockIdx.y % N_HEADS;

    const _Float16* W; const float* bias; _Float16* Out;
    if (blockIdx.z == 0)      { W = Wq; bias = bq; Out = Q; }
    else if (blockIdx.z == 1) { W = Wk; bias = bk; Out = K; }
    else                      { W = Wv; bias = bv; Out = V; }

    const _Float16* Xp = X + ((size_t)b * SEQ + srow0) * E_DIM;
    const _Float16* Wp = W + (size_t)h * E_DIM * H_DIM;
    const float*    bp = bias + h * H_DIM;
    _Float16*       Op = Out + (((size_t)b * N_HEADS + h) * SEQ + srow0) * H_DIM;

    __shared__ _Float16 sA[2][256 * 40];    // 256 rows x 32 K (row-major)
    __shared__ _Float16 sBT[2][64 * 40];    // 64 N x 32 K (transposed!)

    auto issue_A = [&](int buf, int k0) {
        #pragma unroll
        for (int j = 0; j < 4; ++j) {
            int idx = tid + j * 256;
            int r = idx >> 2, q = idx & 3;
            copy_b128_to_lds(Xp + (size_t)r * E_DIM + k0 + q * 8,
                             &sA[buf][r * 40 + q * 8]);
        }
    };
    uint4 rb;
    auto load_rb = [&](int k0) {
        int kr = tid >> 3, q = tid & 7;
        rb = *reinterpret_cast<const uint4*>(Wp + (size_t)(k0 + kr) * H_DIM + q * 8);
    };
    auto store_rb = [&](int buf) {
        U4H8 t; t.u = rb;
        int kr = tid >> 3, q = tid & 7;
        #pragma unroll
        for (int j = 0; j < 8; ++j) sBT[buf][(q * 8 + j) * 40 + kr] = t.h[j];
    };

    v8f acc[2][4] = {};
    const int nk = E_DIM / 32;
    issue_A(0, 0);
    load_rb(0);
    store_rb(0);
    int hf = lane >> 4, ln = lane & 15;

    for (int kc = 0; kc < nk; ++kc) {
        int cur = kc & 1;
        wait_async_copies();                  // my chunk-kc DMA landed
        __syncthreads();                      // everyone's landed; cur^1 readers retired
        if (kc + 1 < nk) {
            issue_A(cur ^ 1, (kc + 1) * 32);  // overlaps with compute below
            load_rb((kc + 1) * 32);
        }

        v16h afr[2];
        #pragma unroll
        for (int mt = 0; mt < 2; ++mt) {
            int m = wave * 32 + mt * 16 + ln;
            #pragma unroll
            for (int e = 0; e < 16; ++e) afr[mt][e] = sA[cur][m * 40 + a_frag_k(e, hf)];
        }
        #pragma unroll
        for (int nt = 0; nt < 4; ++nt) {
            v16h bf;
            #pragma unroll
            for (int e = 0; e < 16; ++e)
                bf[e] = sBT[cur][(nt * 16 + ln) * 40 + 16 * hf + e];   // contiguous
            acc[0][nt] = wmma_f16(afr[0], bf, acc[0][nt]);
            acc[1][nt] = wmma_f16(afr[1], bf, acc[1][nt]);
        }
        if (kc + 1 < nk) store_rb(cur ^ 1);
    }

    #pragma unroll
    for (int nt = 0; nt < 4; ++nt) {
        float bb = bp[nt * 16 + ln];
        #pragma unroll
        for (int mt = 0; mt < 2; ++mt)
            #pragma unroll
            for (int r = 0; r < 8; ++r) {
                int row = wave * 32 + mt * 16 + r + 8 * hf;
                Op[(size_t)row * H_DIM + nt * 16 + ln] =
                    (_Float16)(acc[mt][nt][r] + bb);
            }
    }
}

// ---------------------------------------------------------------------------
// Flash attention: per (b,h), 128 query rows per block, 32-key chunks.
// K tile: async DMA global->LDS (double-buffered). V tile: register path
// with transpose scatter. Q staged through LDS once.
// grid: (SEQ/128, BATCH*N_HEADS), block: 256
// ---------------------------------------------------------------------------
__global__ __launch_bounds__(256, 1) void attn_kernel(
    const _Float16* __restrict__ Q, const _Float16* __restrict__ K,
    const _Float16* __restrict__ V,                   // [B,H,S,D]
    _Float16* __restrict__ cat)                       // [B,S,H*D]
{
    int tid = threadIdx.x, wave = tid >> 5, lane = tid & 31;
    int b = blockIdx.y / N_HEADS, h = blockIdx.y % N_HEADS;
    int q0 = blockIdx.x * 128;
    size_t base = ((size_t)b * N_HEADS + h) * SEQ;
    const _Float16* Qb = Q + (base + q0) * H_DIM;
    const _Float16* Kp = K + base * H_DIM;
    const _Float16* Vp = V + base * H_DIM;

    __shared__ _Float16 sQ[128 * 72];       // 128 rows x 64 D
    __shared__ _Float16 sK[2][32 * 72];     // 32 T x 64 D (row-major)
    __shared__ _Float16 sVT[2][64 * 40];    // 64 D x 32 T (transposed!)
    __shared__ float    sS[8][16][32];
    __shared__ _Float16 sP[8][16][32];
    __shared__ float    sM[8][16], sL[8][16], sC[8][16];

    int hf = lane >> 4, ln = lane & 15;

    // stage Q (coalesced b128)
    #pragma unroll
    for (int j = 0; j < 4; ++j) {
        int idx = tid + j * 256;
        int r = idx >> 3, q = idx & 7;
        *reinterpret_cast<uint4*>(&sQ[r * 72 + q * 8]) =
            *reinterpret_cast<const uint4*>(Qb + (size_t)r * H_DIM + q * 8);
    }

    auto issue_K = [&](int buf, int t0) {
        int r = tid >> 3, q = tid & 7;
        copy_b128_to_lds(Kp + (size_t)(t0 + r) * H_DIM + q * 8,
                         &sK[buf][r * 72 + q * 8]);
    };
    uint4 rv;
    auto load_rv = [&](int t0) {
        int r = tid >> 3, q = tid & 7;
        rv = *reinterpret_cast<const uint4*>(Vp + (size_t)(t0 + r) * H_DIM + q * 8);
    };
    auto store_rv = [&](int buf) {
        U4H8 t; t.u = rv;
        int r = tid >> 3, q = tid & 7;
        #pragma unroll
        for (int j = 0; j < 8; ++j) sVT[buf][(q * 8 + j) * 40 + r] = t.h[j];
    };

    if (lane < 16) { sM[wave][lane] = -1e30f; sL[wave][lane] = 0.0f; }
    issue_K(0, 0);
    load_rv(0);
    store_rv(0);
    __syncthreads();

    // Q A-fragments from LDS (contiguous 8-half runs -> b128 merges)
    v16h qa0, qa1;
    {
        int rowoff = (wave * 16 + ln) * 72;
        #pragma unroll
        for (int e = 0; e < 16; ++e) {
            int kk = a_frag_k(e, hf);
            qa0[e] = sQ[rowoff + kk];
            qa1[e] = sQ[rowoff + 32 + kk];
        }
    }

    v8f acc[4] = {};
    const int ntc = SEQ / 32;
    for (int tc = 0; tc < ntc; ++tc) {
        int cur = tc & 1;
        wait_async_copies();
        __syncthreads();                             // buf[cur] ready everywhere
        if (tc + 1 < ntc) {
            issue_K(cur ^ 1, (tc + 1) * 32);         // overlaps with compute
            load_rv((tc + 1) * 32);
        }

        // scores: two 16x16 tiles, K-dim 64 split into 2 WMMAs each
        #pragma unroll
        for (int tt = 0; tt < 2; ++tt) {
            v16h bk0, bk1;
            int rowoff = (tt * 16 + ln) * 72 + 16 * hf;
            #pragma unroll
            for (int e = 0; e < 16; ++e) {
                bk0[e] = sK[cur][rowoff + e];        // contiguous
                bk1[e] = sK[cur][rowoff + 32 + e];   // contiguous
            }
            v8f s = {};
            s = wmma_f16(qa0, bk0, s);
            s = wmma_f16(qa1, bk1, s);
            #pragma unroll
            for (int r = 0; r < 8; ++r)
                sS[wave][r + 8 * hf][tt * 16 + ln] = s[r];
        }
        __syncthreads();

        // online softmax: one query row per lane (lanes 0..15);
        // 1/sqrt(64) folded into the exponent.
        if (lane < 16) {
            float mold = sM[wave][lane];
            float mmax = mold;
            float vbuf[32];
            #pragma unroll
            for (int j = 0; j < 32; ++j) {
                vbuf[j] = sS[wave][lane][j];
                mmax = fmaxf(mmax, vbuf[j]);
            }
            float scale = __expf((mold - mmax) * 0.125f);
            float sum = 0.0f;
            #pragma unroll
            for (int j = 0; j < 32; ++j) {
                float p = __expf((vbuf[j] - mmax) * 0.125f);
                sum += p;
                sP[wave][lane][j] = (_Float16)p;
            }
            sM[wave][lane] = mmax;
            sL[wave][lane] = sL[wave][lane] * scale + sum;
            sC[wave][lane] = scale;
        }
        __syncthreads();

        // rescale running accumulators
        #pragma unroll
        for (int r = 0; r < 8; ++r) {
            float sc = sC[wave][r + 8 * hf];
            #pragma unroll
            for (int j = 0; j < 4; ++j) acc[j][r] *= sc;
        }

        // O += P @ V : P A-frag (merges to b128), V B-frags from transposed tile
        v16h pa;
        #pragma unroll
        for (int e = 0; e < 16; ++e) pa[e] = sP[wave][ln][a_frag_k(e, hf)];
        #pragma unroll
        for (int nt = 0; nt < 4; ++nt) {
            v16h bvf;
            int rowoff = (nt * 16 + ln) * 40 + 16 * hf;
            #pragma unroll
            for (int e = 0; e < 16; ++e) bvf[e] = sVT[cur][rowoff + e];  // contiguous
            acc[nt] = wmma_f16(pa, bvf, acc[nt]);
        }

        if (tc + 1 < ntc) store_rv(cur ^ 1);
    }
    __syncthreads();

    // finalize: divide by row sums, write head-concatenated f16 [B,S,H*D]
    _Float16* Cp = cat + ((size_t)b * SEQ + q0 + wave * 16) * (N_HEADS * H_DIM)
                       + h * H_DIM;
    #pragma unroll
    for (int r = 0; r < 8; ++r) {
        int row = r + 8 * hf;
        float inv = 1.0f / sL[wave][row];
        #pragma unroll
        for (int nt = 0; nt < 4; ++nt)
            Cp[(size_t)row * (N_HEADS * H_DIM) + nt * 16 + ln] =
                (_Float16)(acc[nt][r] * inv);
    }
}

// ---------------------------------------------------------------------------
// Output projection: out[8192,1024] = cat[8192,1024] @ Wo[1024,1024] + bo
// grid: (8192/256, 1024/64), block: 256; same scheme as qkv_proj.
// ---------------------------------------------------------------------------
__global__ __launch_bounds__(256, 1) void out_proj_kernel(
    const _Float16* __restrict__ Xc, const _Float16* __restrict__ Wo,
    const float* __restrict__ bo, float* __restrict__ Out)
{
    int tid = threadIdx.x, wave = tid >> 5, lane = tid & 31;
    int m0 = blockIdx.x * 256;
    int n0 = blockIdx.y * 64;
    const int Kd = N_HEADS * H_DIM;   // 1024
    const int Nd = E_DIM;             // 1024

    __shared__ _Float16 sA[2][256 * 40];
    __shared__ _Float16 sBT[2][64 * 40];

    auto issue_A = [&](int buf, int k0) {
        #pragma unroll
        for (int j = 0; j < 4; ++j) {
            int idx = tid + j * 256;
            int r = idx >> 2, q = idx & 3;
            copy_b128_to_lds(Xc + (size_t)(m0 + r) * Kd + k0 + q * 8,
                             &sA[buf][r * 40 + q * 8]);
        }
    };
    uint4 rb;
    auto load_rb = [&](int k0) {
        int kr = tid >> 3, q = tid & 7;
        rb = *reinterpret_cast<const uint4*>(Wo + (size_t)(k0 + kr) * Nd + n0 + q * 8);
    };
    auto store_rb = [&](int buf) {
        U4H8 t; t.u = rb;
        int kr = tid >> 3, q = tid & 7;
        #pragma unroll
        for (int j = 0; j < 8; ++j) sBT[buf][(q * 8 + j) * 40 + kr] = t.h[j];
    };

    v8f acc[2][4] = {};
    const int nk = Kd / 32;
    issue_A(0, 0);
    load_rb(0);
    store_rb(0);
    int hf = lane >> 4, ln = lane & 15;

    for (int kc = 0; kc < nk; ++kc) {
        int cur = kc & 1;
        wait_async_copies();
        __syncthreads();
        if (kc + 1 < nk) {
            issue_A(cur ^ 1, (kc + 1) * 32);
            load_rb((kc + 1) * 32);
        }

        v16h afr[2];
        #pragma unroll
        for (int mt = 0; mt < 2; ++mt) {
            int m = wave * 32 + mt * 16 + ln;
            #pragma unroll
            for (int e = 0; e < 16; ++e) afr[mt][e] = sA[cur][m * 40 + a_frag_k(e, hf)];
        }
        #pragma unroll
        for (int nt = 0; nt < 4; ++nt) {
            v16h bf;
            #pragma unroll
            for (int e = 0; e < 16; ++e)
                bf[e] = sBT[cur][(nt * 16 + ln) * 40 + 16 * hf + e];
            acc[0][nt] = wmma_f16(afr[0], bf, acc[0][nt]);
            acc[1][nt] = wmma_f16(afr[1], bf, acc[1][nt]);
        }
        if (kc + 1 < nk) store_rb(cur ^ 1);
    }

    #pragma unroll
    for (int nt = 0; nt < 4; ++nt) {
        float bb = bo[n0 + nt * 16 + ln];
        #pragma unroll
        for (int mt = 0; mt < 2; ++mt)
            #pragma unroll
            for (int r = 0; r < 8; ++r) {
                int row = m0 + wave * 32 + mt * 16 + r + 8 * hf;
                Out[(size_t)row * Nd + n0 + nt * 16 + ln] = acc[mt][nt][r] + bb;
            }
    }
}

// ---------------------------------------------------------------------------
// Host entry
// ---------------------------------------------------------------------------
extern "C" void kernel_launch(void* const* d_in, const int* in_sizes, int n_in,
                              void* d_out, int out_size, void* d_ws, size_t ws_size,
                              hipStream_t stream) {
    const float* x   = (const float*)d_in[0];
    const float* Wq  = (const float*)d_in[1];
    const float* bq  = (const float*)d_in[2];
    const float* Wk  = (const float*)d_in[3];
    const float* bk  = (const float*)d_in[4];
    const float* Wv  = (const float*)d_in[5];
    const float* bv  = (const float*)d_in[6];
    const float* Wo  = (const float*)d_in[7];
    const float* bo  = (const float*)d_in[8];
    float* out = (float*)d_out;

    const size_t nX   = (size_t)BATCH * SEQ * E_DIM;            // 8388608
    const size_t nW   = (size_t)N_HEADS * E_DIM * H_DIM;        // 1048576
    const size_t nWo  = (size_t)E_DIM * E_DIM;                  // 1048576
    const size_t nQKV = (size_t)BATCH * N_HEADS * SEQ * H_DIM;  // 8388608

    char* ws = (char*)d_ws;
    _Float16* hX   = (_Float16*)ws; ws += nX * 2;
    _Float16* hWq  = (_Float16*)ws; ws += nW * 2;
    _Float16* hWk  = (_Float16*)ws; ws += nW * 2;
    _Float16* hWv  = (_Float16*)ws; ws += nW * 2;
    _Float16* hWo  = (_Float16*)ws; ws += nWo * 2;
    _Float16* hQ   = (_Float16*)ws; ws += nQKV * 2;
    _Float16* hK   = (_Float16*)ws; ws += nQKV * 2;
    _Float16* hV   = (_Float16*)ws; ws += nQKV * 2;
    _Float16* hCat = (_Float16*)ws; ws += nX * 2;

    cvt_f32_f16_kernel<<<2048, 256, 0, stream>>>((const float4*)x,  (uint2*)hX,  (int)(nX / 4));
    cvt_f32_f16_kernel<<<512,  256, 0, stream>>>((const float4*)Wq, (uint2*)hWq, (int)(nW / 4));
    cvt_f32_f16_kernel<<<512,  256, 0, stream>>>((const float4*)Wk, (uint2*)hWk, (int)(nW / 4));
    cvt_f32_f16_kernel<<<512,  256, 0, stream>>>((const float4*)Wv, (uint2*)hWv, (int)(nW / 4));
    cvt_f32_f16_kernel<<<512,  256, 0, stream>>>((const float4*)Wo, (uint2*)hWo, (int)(nWo / 4));

    qkv_proj_kernel<<<dim3(SEQ / 256, BATCH * N_HEADS, 3), 256, 0, stream>>>(
        hX, hWq, hWk, hWv, bq, bk, bv, hQ, hK, hV);

    attn_kernel<<<dim3(SEQ / 128, BATCH * N_HEADS), 256, 0, stream>>>(
        hQ, hK, hV, hCat);

    out_proj_kernel<<<dim3((BATCH * SEQ) / 256, E_DIM / 64), 256, 0, stream>>>(
        hCat, hWo, bo, out);
}
